// MSA4_68624987455589
// MI455X (gfx1250) — matrix operand order repaired and checked by reading, not compiled, not run on hardware
//
#include <hip/hip_runtime.h>
#include <stdint.h>

// ---------------------------------------------------------------------------
// CDNA5 (gfx1250) wave32 WMMA types
// ---------------------------------------------------------------------------
typedef __attribute__((ext_vector_type(16))) __bf16 v16bf;
typedef __attribute__((ext_vector_type(8)))  __bf16 v8bf;
typedef __attribute__((ext_vector_type(8)))  float  v8f;

__device__ __forceinline__ float gelu_f(float x) {
  return 0.5f * x * (1.0f + tanhf(0.79788456080286541f * (x + 0.044715f * x * x * x)));
}

// Async global -> LDS copy (CDNA5, tracked on ASYNCcnt)
__device__ __forceinline__ void async_copy_b128(const void* gptr, void* lptr) {
  uint32_t loff = (uint32_t)(uintptr_t)lptr;   // addrspace(3) offset in low 32 bits
  asm volatile("global_load_async_to_lds_b128 %0, %1, off"
               :: "v"(loff), "v"(gptr) : "memory");
}
__device__ __forceinline__ void wait_async_le4() { asm volatile("s_wait_asynccnt 4" ::: "memory"); }
__device__ __forceinline__ void wait_async_le2() { asm volatile("s_wait_asynccnt 2" ::: "memory"); }
__device__ __forceinline__ void wait_async_0()  { asm volatile("s_wait_asynccnt 0" ::: "memory"); }

// ---------------------------------------------------------------------------
// Shared WMMA tile step: one BK=32 slab. Wave computes 64x32 (4x2 16x16 tiles).
// A LDS tile: [128][32] bf16 row-major (M x K); B LDS tile: [128][32] (N x K).
// A frag (16x32): lane = g*16+mr holds row mr, K-chunks [g*8, g*8+8) and
// [16+g*8, ...). B frag (32x16): lane holds col mr, same K-chunks as A.
// ---------------------------------------------------------------------------
__device__ __forceinline__ void wmma_tile_step(const __bf16 (*sA)[32], const __bf16 (*sB)[32],
                                               int wm, int wn, int lane, v8f acc[4][2]) {
  const int mr = lane & 15;
  const int g  = lane >> 4;
  v16bf af[4];
  v16bf bfr[2];
#pragma unroll
  for (int mt = 0; mt < 4; ++mt) {
    const v8bf* pa = reinterpret_cast<const v8bf*>(&sA[wm * 64 + mt * 16 + mr][g * 8]);
    v8bf lo = pa[0];
    v8bf hi = pa[2];   // +16 elements
#pragma unroll
    for (int i = 0; i < 8; ++i) { af[mt][i] = lo[i]; af[mt][i + 8] = hi[i]; }
  }
#pragma unroll
  for (int nt = 0; nt < 2; ++nt) {
    const v8bf* pb = reinterpret_cast<const v8bf*>(&sB[wn * 32 + nt * 16 + mr][g * 8]);
    v8bf lo = pb[0];
    v8bf hi = pb[2];
#pragma unroll
    for (int i = 0; i < 8; ++i) { bfr[nt][i] = lo[i]; bfr[nt][i + 8] = hi[i]; }
  }
#pragma unroll
  for (int mt = 0; mt < 4; ++mt)
#pragma unroll
    for (int nt = 0; nt < 2; ++nt)
      acc[mt][nt] = __builtin_amdgcn_wmma_f32_16x16x32_bf16(
          false, af[mt], false, bfr[nt], (short)0, acc[mt][nt], false, false);
}

// ---------------------------------------------------------------------------
// Generic batched bf16 GEMM:  C[z][M][N] = A[z][M][K] * Bt[z][N][K]^T
// EPI bits: 1=bias, 2=residual(f32), 4=gelu, 8=bf16 out, 16=scale
// z decomposes as (zb, zh) with zh = z % ZH. All strides in elements.
// ---------------------------------------------------------------------------
template <int EPI>
__global__ __launch_bounds__(256) void gemm_bf16_kernel(
    const __bf16* __restrict__ A, long sAb, long sAh, int lda,
    const __bf16* __restrict__ Bt, long sBb, long sBh, int ldb,
    void* __restrict__ Cv, long sCb, long sCh, int ldc,
    const float* __restrict__ bias,
    const float* __restrict__ resid, long sRb, long sRh,
    float scale, int ZH, int K) {
  __shared__ __align__(16) __bf16 sA[2][128][32];
  __shared__ __align__(16) __bf16 sB[2][128][32];
  const int tid = threadIdx.x, lane = tid & 31, wid = tid >> 5;
  const int wm = wid & 1, wn = wid >> 1;
  const int zb = blockIdx.z / ZH, zh = blockIdx.z % ZH;

  const __bf16* Az = A + zb * sAb + zh * sAh + (long)blockIdx.y * 128 * lda;
  const __bf16* Bz = Bt + zb * sBb + zh * sBh + (long)blockIdx.x * 128 * ldb;

  auto stage = [&](int buf, int k0) {
#pragma unroll
    for (int q = 0; q < 2; ++q) {
      int e = tid * 2 + q;
      int row = e >> 2, col = (e & 3) * 8;
      async_copy_b128(Az + (long)row * lda + k0 + col, &sA[buf][row][col]);
      async_copy_b128(Bz + (long)row * ldb + k0 + col, &sB[buf][row][col]);
    }
  };

  v8f acc[4][2] = {};
  const int nk = K >> 5;
  stage(0, 0);
  for (int ki = 0; ki < nk; ++ki) {
    const int buf = ki & 1;
    if (ki + 1 < nk) { stage(buf ^ 1, (ki + 1) * 32); wait_async_le4(); }
    else             { wait_async_0(); }
    __syncthreads();
    wmma_tile_step(sA[buf], sB[buf], wm, wn, lane, acc);
    __syncthreads();
  }

  // Epilogue: C/D layout -> row = tile + g*8 + r, col = tile + (lane&15)
  float*  Cf = (float*)Cv;
  __bf16* Cb = (__bf16*)Cv;
  const int g = lane >> 4, cn = lane & 15;
  const long czoff = zb * sCb + zh * sCh;
#pragma unroll
  for (int mt = 0; mt < 4; ++mt)
#pragma unroll
    for (int nt = 0; nt < 2; ++nt)
#pragma unroll
      for (int r = 0; r < 8; ++r) {
        const int row = blockIdx.y * 128 + wm * 64 + mt * 16 + g * 8 + r;
        const int col = blockIdx.x * 128 + wn * 32 + nt * 16 + cn;
        float v = acc[mt][nt][r];
        if (EPI & 16) v *= scale;
        if (EPI & 1)  v += bias[col];
        if (EPI & 2)  v += resid[zb * sRb + zh * sRh + (long)row * ldc + col];
        if (EPI & 4)  v = gelu_f(v);
        const long ci = czoff + (long)row * ldc + col;
        if (EPI & 8) Cb[ci] = (__bf16)v; else Cf[ci] = v;
      }
}

// ---------------------------------------------------------------------------
// 3x3 conv (SAME) on NHWC 8x32x32x2048 as implicit-GEMM with a shared halo
// tile: one 6x34x32 input halo slab per K-chunk serves all 9 taps; weight
// tiles stream through a 2-deep async LDS pipeline.
// X: [8192][2048] bf16, Wt: [9][Co=2048][Ci=2048] bf16, out = gelu(conv+bias)
// ---------------------------------------------------------------------------
__global__ __launch_bounds__(256) void conv3x3_kernel(
    const __bf16* __restrict__ X, const __bf16* __restrict__ Wt,
    const float* __restrict__ bias, __bf16* __restrict__ Y) {
  // halo: [2][6][34][32] bf16 (y0-1 .. y0+4) x (x=-1 .. 32) x 32 channels
  __shared__ __align__(16) __bf16 sH[2][6 * 34 * 32];
  __shared__ __align__(16) __bf16 sB[2][128][32];
  const int tid = threadIdx.x, lane = tid & 31, wid = tid >> 5;
  const int wm = wid & 1, wn = wid >> 1;
  const int bn = blockIdx.x;            // N/128
  const int bm = blockIdx.y;            // M/128 = 4 image rows of one batch
  const int pb = bm >> 3;               // batch
  const int y0 = (bm & 7) * 4;          // first image row of this block
  const long xbase = ((long)pb * 1024) * 2048;

  // ---- synchronous halo stage (zero-filled padding), 816 16B chunks ----
  auto stage_halo = [&](int hb, int kc) {
#pragma unroll
    for (int q = 0; q < 4; ++q) {
      int e = tid + q * 256;
      if (e < 816) {
        int cc = e & 3;                 // channel chunk (8 bf16)
        int pix = e >> 2;               // yy*34 + xx, 0..203
        int yy = pix / 34, xx = pix - yy * 34;
        int gy = y0 + yy - 1, gx = xx - 1;
        int4 v = make_int4(0, 0, 0, 0);
        if (gy >= 0 && gy < 32 && gx >= 0 && gx < 32)
          v = *(const int4*)(X + xbase + ((long)(gy * 32 + gx)) * 2048 + kc * 32 + cc * 8);
        *(int4*)&sH[hb][(yy * 34 + xx) * 32 + cc * 8] = v;
      }
    }
  };
  // ---- async weight stage: tap t, K-chunk kc -> sB[buf] (2 ops/thread) ----
  auto stage_w = [&](int buf, int t, int kc) {
    const __bf16* W = Wt + (long)t * 2048 * 2048 + (long)(bn * 128) * 2048 + kc * 32;
#pragma unroll
    for (int q = 0; q < 2; ++q) {
      int e = tid * 2 + q;
      int row = e >> 2, col = (e & 3) * 8;
      async_copy_b128(W + (long)row * 2048 + col, &sB[buf][row][col]);
    }
  };

  v8f acc[4][2] = {};
  const int mr = lane & 15, g = lane >> 4;

  stage_halo(0, 0);
  stage_w(0, 0, 0);
  const int nstage = 64 * 9;
  for (int s = 0; s < nstage; ++s) {
    const int kc = s / 9, t = s - kc * 9;
    const int hb = kc & 1, bb = s & 1;
    if (s + 1 < nstage) {
      const int kn = (s + 1) / 9, tn = (s + 1) - kn * 9;
      stage_w(bb ^ 1, tn, kn);
      if (t == 8) stage_halo(hb ^ 1, kc + 1);
      wait_async_le2();
    } else {
      wait_async_0();
    }
    __syncthreads();

    // ---- compute tap t from halo + sB[bb] ----
    {
      const __bf16* H = sH[hb];
      const int ty = t / 3, tx = t - ty * 3;
      v16bf af[4];
      v16bf bfr[2];
#pragma unroll
      for (int mt = 0; mt < 4; ++mt) {
        const int m = wm * 64 + mt * 16 + mr;
        const int y = m >> 5, x = m & 31;
        const __bf16* base = H + ((y + ty) * 34 + (x + tx)) * 32 + g * 8;
        const v8bf* pa = reinterpret_cast<const v8bf*>(base);
        v8bf lo = pa[0];
        v8bf hi = pa[2];   // +16 channels
#pragma unroll
        for (int i = 0; i < 8; ++i) { af[mt][i] = lo[i]; af[mt][i + 8] = hi[i]; }
      }
#pragma unroll
      for (int nt = 0; nt < 2; ++nt) {
        const v8bf* pbv = reinterpret_cast<const v8bf*>(&sB[bb][wn * 32 + nt * 16 + mr][g * 8]);
        v8bf lo = pbv[0];
        v8bf hi = pbv[2];
#pragma unroll
        for (int i = 0; i < 8; ++i) { bfr[nt][i] = lo[i]; bfr[nt][i + 8] = hi[i]; }
      }
#pragma unroll
      for (int mt = 0; mt < 4; ++mt)
#pragma unroll
        for (int nt = 0; nt < 2; ++nt)
          acc[mt][nt] = __builtin_amdgcn_wmma_f32_16x16x32_bf16(
              false, af[mt], false, bfr[nt], (short)0, acc[mt][nt], false, false);
    }
    __syncthreads();
  }

  const int cn = lane & 15;
#pragma unroll
  for (int mt = 0; mt < 4; ++mt)
#pragma unroll
    for (int nt = 0; nt < 2; ++nt)
#pragma unroll
      for (int r = 0; r < 8; ++r) {
        const int row = bm * 128 + wm * 64 + mt * 16 + g * 8 + r;
        const int col = bn * 128 + wn * 32 + nt * 16 + cn;
        float v = acc[mt][nt][r] + bias[col];
        Y[(long)row * 2048 + col] = (__bf16)gelu_f(v);
      }
}

// ---------------------------------------------------------------------------
// Elementwise / reduction helpers
// ---------------------------------------------------------------------------
__global__ void k_f2bf(const float* __restrict__ in, __bf16* __restrict__ out, long n) {
  long i = (long)blockIdx.x * 256 + threadIdx.x;
  if (i < n) out[i] = (__bf16)in[i];
}

__global__ void k_nchw2nhwc(const float* __restrict__ x, __bf16* __restrict__ y) {
  long i = (long)blockIdx.x * 256 + threadIdx.x;          // 8*4096*2048
  int c = (int)(i & 2047);
  int hw = (int)((i >> 11) & 4095);
  int b = (int)(i >> 23);
  y[i] = (__bf16)x[((long)b * 2048 + c) * 4096 + hw];
}

__global__ void k_wtaps(const float* __restrict__ w, __bf16* __restrict__ wt) {
  long i = (long)blockIdx.x * 256 + threadIdx.x;          // 9*2048*2048
  int t = (int)(i >> 22);
  long rem = i & 4194303;                                  // o*2048 + ci
  wt[i] = (__bf16)w[rem * 9 + t];
}

__global__ __launch_bounds__(256) void k_bn_stats(const float* __restrict__ X,
                                                  float* __restrict__ st, int rows, int C) {
  const int c = blockIdx.x, tid = threadIdx.x;
  float s = 0.f, s2 = 0.f;
  for (int r = tid; r < rows; r += 256) {
    float v = X[(long)r * C + c];
    s += v; s2 += v * v;
  }
  __shared__ float sh0[256], sh1[256];
  sh0[tid] = s; sh1[tid] = s2; __syncthreads();
  for (int o = 128; o > 0; o >>= 1) {
    if (tid < o) { sh0[tid] += sh0[tid + o]; sh1[tid] += sh1[tid + o]; }
    __syncthreads();
  }
  if (tid == 0) {
    float mu = sh0[0] / rows;
    float var = sh1[0] / rows - mu * mu;
    st[c * 2] = mu;
    st[c * 2 + 1] = rsqrtf(var + 1e-5f);
  }
}

// BN + relu + unfold3x3 stride-2 (NHWC in, k = c*9 + patch out), bf16 out
__global__ void k_unfold(const float* __restrict__ h1, const float* __restrict__ st,
                         const float* __restrict__ g, const float* __restrict__ b_,
                         __bf16* __restrict__ u) {
  long idx = (long)blockIdx.x * 256 + threadIdx.x;        // 8*32*32*512
  int c = (int)(idx & 511);
  int wo = (int)((idx >> 9) & 31);
  int ho = (int)((idx >> 14) & 31);
  int bb = (int)(idx >> 19);
  float mu = st[c * 2], rs = st[c * 2 + 1];
  float sc = rs * g[c];
  float sh = b_[c] - mu * sc;
  long ubase = ((long)((bb * 32 + ho) * 32 + wo)) * 4608 + c * 9;
#pragma unroll
  for (int p = 0; p < 9; ++p) {
    int yi = 2 * ho + p / 3 - 1, xi = 2 * wo + p % 3 - 1;
    float v = 0.f;
    if (yi >= 0 && yi < 64 && xi >= 0 && xi < 64) {
      v = h1[((long)((bb * 64 + yi) * 64 + xi)) * 512 + c];
      v = fmaxf(v * sc + sh, 0.f);
    }
    u[ubase + p] = (__bf16)v;
  }
}

// Fused instance norm: per (b, c) stats over 1024 rows, writes bf16
__global__ __launch_bounds__(256) void k_in_norm(const float* __restrict__ X,
                                                 __bf16* __restrict__ Y,
                                                 const float* __restrict__ g,
                                                 const float* __restrict__ b_) {
  const int bc = blockIdx.x;          // 8*512
  const int bb = bc >> 9, c = bc & 511;
  const int tid = threadIdx.x;
  const float* base = X + ((long)bb * 1024) * 512 + c;
  float s = 0.f, s2 = 0.f;
  for (int r = tid; r < 1024; r += 256) {
    float v = base[(long)r * 512];
    s += v; s2 += v * v;
  }
  __shared__ float sh0[256], sh1[256];
  __shared__ float scs, shs;
  sh0[tid] = s; sh1[tid] = s2; __syncthreads();
  for (int o = 128; o > 0; o >>= 1) {
    if (tid < o) { sh0[tid] += sh0[tid + o]; sh1[tid] += sh1[tid + o]; }
    __syncthreads();
  }
  if (tid == 0) {
    float mu = sh0[0] * (1.f / 1024.f);
    float var = sh1[0] * (1.f / 1024.f) - mu * mu;
    float rs = rsqrtf(var + 1e-5f);
    scs = rs * g[c];
    shs = b_[c] - mu * rs * g[c];
  }
  __syncthreads();
  float sc = scs, shv = shs;
  __bf16* yb = Y + ((long)bb * 1024) * 512 + c;
  for (int r = tid; r < 1024; r += 256)
    yb[(long)r * 512] = (__bf16)(base[(long)r * 512] * sc + shv);
}

__global__ __launch_bounds__(256) void k_softmax(__bf16* __restrict__ S) {
  const long row = blockIdx.x;        // 32*1024 rows of length 1024
  __bf16* p = S + (row << 10);
  const int tid = threadIdx.x;
  float v[4];
  float mx = -3.0e38f;
#pragma unroll
  for (int i = 0; i < 4; ++i) { v[i] = (float)p[tid + i * 256]; mx = fmaxf(mx, v[i]); }
  __shared__ float red[256];
  red[tid] = mx; __syncthreads();
  for (int o = 128; o > 0; o >>= 1) { if (tid < o) red[tid] = fmaxf(red[tid], red[tid + o]); __syncthreads(); }
  mx = red[0]; __syncthreads();
  float s = 0.f;
#pragma unroll
  for (int i = 0; i < 4; ++i) { v[i] = expf(v[i] - mx); s += v[i]; }
  red[tid] = s; __syncthreads();
  for (int o = 128; o > 0; o >>= 1) { if (tid < o) red[tid] += red[tid + o]; __syncthreads(); }
  const float inv = 1.f / red[0];
#pragma unroll
  for (int i = 0; i < 4; ++i) p[tid + i * 256] = (__bf16)(v[i] * inv);
}

// V transpose: vt[b][h][d][i] = qkv[b*1024+i][1024 + h*128 + d]
__global__ void k_vt(const __bf16* __restrict__ qkv, __bf16* __restrict__ vt) {
  long idx = (long)blockIdx.x * 256 + threadIdx.x;        // 8*4*128*1024
  int i = (int)(idx & 1023);
  int d = (int)((idx >> 10) & 127);
  int h = (int)((idx >> 17) & 3);
  int b = (int)(idx >> 19);
  vt[idx] = qkv[((long)(b * 1024 + i)) * 1536 + 1024 + h * 128 + d];
}

// align-corners bilinear 32x32 -> 64x64; NHWC f32 in, NCHW f32 out
__global__ void k_bilinear(const float* __restrict__ h3, float* __restrict__ out) {
  long idx = (long)blockIdx.x * 256 + threadIdx.x;        // 8*512*64*64
  int x = (int)(idx & 63);
  int y = (int)((idx >> 6) & 63);
  int c = (int)((idx >> 12) & 511);
  int b = (int)(idx >> 21);
  float sy = y * (31.0f / 63.0f), sx = x * (31.0f / 63.0f);
  int y0 = (int)floorf(sy), x0 = (int)floorf(sx);
  int y1 = min(y0 + 1, 31), x1 = min(x0 + 1, 31);
  float wy = sy - y0, wx = sx - x0;
  long base = (long)b * 1024;
  float v00 = h3[(base + y0 * 32 + x0) * 512 + c];
  float v01 = h3[(base + y0 * 32 + x1) * 512 + c];
  float v10 = h3[(base + y1 * 32 + x0) * 512 + c];
  float v11 = h3[(base + y1 * 32 + x1) * 512 + c];
  float top = v00 * (1.f - wx) + v01 * wx;
  float bot = v10 * (1.f - wx) + v11 * wx;
  out[idx] = top * (1.f - wy) + bot * wy;
}

// ---------------------------------------------------------------------------
// Host side
// ---------------------------------------------------------------------------
static inline void run_gemm(int epi,
                            const __bf16* A, long sAb, long sAh, int lda,
                            const __bf16* Bt, long sBb, long sBh, int ldb,
                            void* C, long sCb, long sCh, int ldc,
                            const float* bias, const float* resid, long sRb, long sRh,
                            float scale, int M, int N, int Z, int ZH, int K,
                            hipStream_t s) {
  dim3 g(N / 128, M / 128, Z), b(256);
#define GEMM_CASE(E) \
  case E: gemm_bf16_kernel<E><<<g, b, 0, s>>>(A, sAb, sAh, lda, Bt, sBb, sBh, ldb, C, sCb, sCh, ldc, \
                                              bias, resid, sRb, sRh, scale, ZH, K); break;
  switch (epi) {
    GEMM_CASE(0) GEMM_CASE(1) GEMM_CASE(2) GEMM_CASE(3)
    GEMM_CASE(8) GEMM_CASE(9) GEMM_CASE(24)
    default: break;
  }
#undef GEMM_CASE
}

extern "C" void kernel_launch(void* const* d_in, const int* in_sizes, int n_in,
                              void* d_out, int out_size, void* d_ws, size_t ws_size,
                              hipStream_t stream) {
  (void)in_sizes; (void)n_in; (void)out_size; (void)ws_size;
  const float* x     = (const float*)d_in[0];
  const float* w_red = (const float*)d_in[1];
  const float* bn_g  = (const float*)d_in[2];
  const float* bn_b  = (const float*)d_in[3];
  const float* w_emb = (const float*)d_in[4];
  const float* b_emb = (const float*)d_in[5];
  const float* ln_g  = (const float*)d_in[6];
  const float* ln_b  = (const float*)d_in[7];
  const float* w_qkv = (const float*)d_in[8];
  const float* w_ao  = (const float*)d_in[9];
  const float* w_f1  = (const float*)d_in[10];
  const float* b_f1  = (const float*)d_in[11];
  const float* w_f2  = (const float*)d_in[12];
  const float* b_f2  = (const float*)d_in[13];
  const float* w_f3  = (const float*)d_in[14];
  const float* b_f3  = (const float*)d_in[15];

  char* ws = (char*)d_ws;
  size_t off = 0;
  auto alloc = [&](size_t bytes) -> void* {
    void* p = ws + off;
    off += (bytes + 255) & ~(size_t)255;
    return p;
  };

  __bf16* xh    = (__bf16*)alloc(134217728UL); // x NHWC bf16
  __bf16* wredb = (__bf16*)alloc(2097152UL);
  float*  h1    = (float*) alloc(67108864UL);  // conv1x1 reduce out, NHWC f32
  float*  bnst  = (float*) alloc(4096UL);
  __bf16* u     = (__bf16*)alloc(75497472UL);  // unfold, [8192][4608]
  __bf16* wembb = (__bf16*)alloc(4718592UL);
  float*  e     = (float*) alloc(16777216UL);  // embed out (residual), [8192][512]
  __bf16* n1    = (__bf16*)alloc(8388608UL);
  __bf16* wqkvb = (__bf16*)alloc(1572864UL);
  __bf16* qkvb  = (__bf16*)alloc(25165824UL);  // [8192][1536]
  __bf16* sim   = (__bf16*)alloc(67108864UL);  // [32][1024][1024]
  __bf16* vt    = (__bf16*)alloc(8388608UL);   // [32][128][1024]
  __bf16* ob    = (__bf16*)alloc(8388608UL);   // attn out pre-proj [8192][512]
  __bf16* waob  = (__bf16*)alloc(524288UL);
  float*  h2    = (float*) alloc(16777216UL);  // post-attn residual f32
  __bf16* n2    = (__bf16*)alloc(8388608UL);
  __bf16* wff1b = (__bf16*)alloc(2097152UL);
  __bf16* f1    = (__bf16*)alloc(33554432UL);  // [8192][2048] bf16
  __bf16* wtaps = (__bf16*)alloc(75497472UL);  // [9][2048][2048] bf16
  __bf16* g2    = (__bf16*)alloc(33554432UL);  // gelu(conv3x3) bf16
  __bf16* wff3b = (__bf16*)alloc(2097152UL);
  float*  h3    = (float*) alloc(16777216UL);  // final NHWC f32

  // ---- weight conversions (f32 -> bf16; weights are already [Co][Ci] = Bt) ----
  k_f2bf<<<(512L * 2048 + 255) / 256, 256, 0, stream>>>(w_red, wredb, 512L * 2048);
  k_f2bf<<<(512L * 4608 + 255) / 256, 256, 0, stream>>>(w_emb, wembb, 512L * 4608);
  k_f2bf<<<(1536L * 512 + 255) / 256, 256, 0, stream>>>(w_qkv, wqkvb, 1536L * 512);
  k_f2bf<<<(512L * 512 + 255) / 256, 256, 0, stream>>>(w_ao, waob, 512L * 512);
  k_f2bf<<<(2048L * 512 + 255) / 256, 256, 0, stream>>>(w_f1, wff1b, 2048L * 512);
  k_f2bf<<<(512L * 2048 + 255) / 256, 256, 0, stream>>>(w_f3, wff3b, 512L * 2048);
  k_wtaps<<<(9L * 4194304) / 256, 256, 0, stream>>>(w_f2, wtaps);

  // ---- stage 1: x NCHW -> NHWC bf16; reduce conv1x1 [32768x512] ----
  k_nchw2nhwc<<<67108864L / 256, 256, 0, stream>>>(x, xh);
  run_gemm(0, xh, 0, 0, 2048, wredb, 0, 0, 2048, h1, 0, 0, 512,
           nullptr, nullptr, 0, 0, 1.f, 32768, 512, 1, 1, 2048, stream);

  // ---- batch norm stats + fused BN/relu/unfold -> u ----
  k_bn_stats<<<512, 256, 0, stream>>>(h1, bnst, 32768, 512);
  k_unfold<<<4194304L / 256, 256, 0, stream>>>(h1, bnst, bn_g, bn_b, u);

  // ---- embed conv1x1 + bias -> e (f32, kept as residual) ----
  run_gemm(1, u, 0, 0, 4608, wembb, 0, 0, 4608, e, 0, 0, 512,
           b_emb, nullptr, 0, 0, 1.f, 8192, 512, 1, 1, 4608, stream);

  // ---- instance norm 1 -> n1; qkv projection ----
  k_in_norm<<<4096, 256, 0, stream>>>(e, n1, ln_g, ln_b);
  run_gemm(8, n1, 0, 0, 512, wqkvb, 0, 0, 512, qkvb, 0, 0, 1536,
           nullptr, nullptr, 0, 0, 1.f, 8192, 1536, 1, 1, 512, stream);

  // ---- attention: sim = Q K^T * d^-0.5 (batched over 8*4 (b,h)) ----
  run_gemm(24, qkvb, 1024L * 1536, 128, 1536,
           qkvb + 512, 1024L * 1536, 128, 1536,
           sim, 4194304L, 1048576L, 1024,
           nullptr, nullptr, 0, 0, 0.08838834764831845f,
           1024, 1024, 32, 4, 128, stream);
  k_softmax<<<32768, 256, 0, stream>>>(sim);
  k_vt<<<4194304L / 256, 256, 0, stream>>>(qkvb, vt);
  // out = P V  -> ob [8192][512]
  run_gemm(8, sim, 4194304L, 1048576L, 1024,
           vt, 524288L, 131072L, 1024,
           ob, 524288L, 128, 512,
           nullptr, nullptr, 0, 0, 1.f, 1024, 128, 32, 4, 1024, stream);
  // output projection + residual e -> h2 (f32)
  run_gemm(2, ob, 0, 0, 512, waob, 0, 0, 512, h2, 0, 0, 512,
           nullptr, e, 0, 0, 1.f, 8192, 512, 1, 1, 512, stream);

  // ---- instance norm 2 -> n2; ff1 + bias -> f1 (bf16) ----
  k_in_norm<<<4096, 256, 0, stream>>>(h2, n2, ln_g, ln_b);
  run_gemm(9, n2, 0, 0, 512, wff1b, 0, 0, 512, f1, 0, 0, 2048,
           b_f1, nullptr, 0, 0, 1.f, 8192, 2048, 1, 1, 512, stream);

  // ---- conv3x3 + bias + gelu -> g2 (bf16) ----
  {
    dim3 g(16, 64), b(256);
    conv3x3_kernel<<<g, b, 0, stream>>>(f1, wtaps, b_f2, g2);
  }

  // ---- ff3 + bias + residual h2 -> h3 (f32 NHWC) ----
  run_gemm(3, g2, 0, 0, 2048, wff3b, 0, 0, 2048, h3, 0, 0, 512,
           b_f3, h2, 0, 0, 1.f, 8192, 512, 1, 1, 2048, stream);

  // ---- bilinear align-corners 32->64, emit NCHW f32 ----
  k_bilinear<<<16777216L / 256, 256, 0, stream>>>(h3, (float*)d_out);
}
